// VectorQuantizer_60507499266080
// MI455X (gfx1250) — compile-verified, matrix operand order checked by reading, not run on hardware
//
#include <hip/hip_runtime.h>
#include <stdint.h>

// Problem constants (from reference)
#define NUM_EMBS 8192
#define EMB      256
#define NPTS     32768                // 32 * 32 * 32 points
#define QOUT     ((size_t)NPTS * EMB) // 8388608 quantized floats, then NPTS index-floats

// WMMA vector types (CDNA5 / gfx1250, wave32)
typedef __attribute__((ext_vector_type(16))) __bf16        v16bf;
typedef __attribute__((ext_vector_type(8)))  float         v8f;
typedef __attribute__((ext_vector_type(4)))  unsigned int  v4u;

union Frag16 {            // 16 bf16 = 8 VGPRs, loadable as two 16B chunks
    v16bf v;
    v4u   q[2];
};

__device__ __forceinline__ unsigned short f32_to_bf16_rne(float f) {
    unsigned int u = __float_as_uint(f);
    u += 0x7fffu + ((u >> 16) & 1u);     // round to nearest even
    return (unsigned short)(u >> 16);
}

// ---------------------------------------------------------------------------
// Kernel 1a: codebook -> bf16, and fp32 row squared norms
// ---------------------------------------------------------------------------
__global__ __launch_bounds__(256)
void cb_prep_kernel(const float* __restrict__ cb,
                    unsigned short* __restrict__ cbB,
                    float* __restrict__ cbSq) {
    const int k = blockIdx.x;
    const int c = threadIdx.x;
    const float v = cb[(size_t)k * EMB + c];
    cbB[(size_t)k * EMB + c] = f32_to_bf16_rne(v);
    float s = v * v;
    #pragma unroll
    for (int off = 16; off; off >>= 1) s += __shfl_xor(s, off, 32);
    __shared__ float sh[8];
    if ((c & 31) == 0) sh[c >> 5] = s;
    __syncthreads();
    if (c == 0) {
        float t = 0.f;
        #pragma unroll
        for (int i = 0; i < 8; ++i) t += sh[i];
        cbSq[k] = t;
    }
}

// ---------------------------------------------------------------------------
// Kernel 1b: x [b,c,h,w] fp32 -> xf [n=(b,h,w), c] bf16, LDS tile transpose
// grid = (8 c-chunks, 32 h, 32 b), block = (32, 32)
// ---------------------------------------------------------------------------
__global__ __launch_bounds__(1024)
void x_transpose_kernel(const float* __restrict__ x,
                        unsigned short* __restrict__ xfB) {
    __shared__ unsigned short sh[32][33];
    const int tx = threadIdx.x;          // w on read, c-in-chunk on write
    const int ty = threadIdx.y;          // c-in-chunk on read, w on write
    const int cc = blockIdx.x;           // 0..7
    const int h  = blockIdx.y;           // 0..31
    const int b  = blockIdx.z;           // 0..31
    const int c  = cc * 32 + ty;
    const float v = x[(((size_t)b * EMB + c) << 10) + h * 32 + tx]; // coalesced in w
    sh[ty][tx] = f32_to_bf16_rne(v);
    __syncthreads();
    const size_t n = (size_t)b * 1024 + h * 32 + ty;
    xfB[n * EMB + cc * 32 + tx] = sh[tx][ty];                       // coalesced in c
}

// ---------------------------------------------------------------------------
// Kernel 2: bf16 WMMA GEMM (cross terms) fused with running row-argmin of
//           d2[k] = cb_sq[k] - 2 * <x_n, cb_k>
// WG = 256 threads (8 waves); wave owns 32 rows (2 x 16-row M blocks).
// B tile (16 cols x 256 C) double-buffered in LDS, shared by all waves.
// B fragments software-pipelined in registers; cbSq pipelined one tile ahead
// so the min-update never drains the staging global loads.
// ---------------------------------------------------------------------------
#define KTILES        (NUM_EMBS / 16)  // 512
#define LDS_STRIDE_BF 264              // 256 + 8 pad -> 4-dword bank stride/row

__global__ __launch_bounds__(256)
void vq_argmin_kernel(const unsigned short* __restrict__ xfB,
                      const unsigned short* __restrict__ cbB,
                      const float* __restrict__ cbSq,
                      int* __restrict__ idxOut) {
    __shared__ unsigned short ldsB[2][16 * LDS_STRIDE_BF];

    const int tid  = threadIdx.x;
    const int lane = tid & 31;
    const int wave = tid >> 5;
    const int l15  = lane & 15;
    const int hsel = (lane >> 4) & 1;           // lane-half selects K sub-range
    const int rBase = blockIdx.x * 256 + wave * 32;

    // Resident A fragments: 2 M-blocks x 8 c-chunks (16x32 bf16 each).
    // 16-bit A layout: lanes 0-15 -> K {0..7,16..23}; lanes 16-31 -> K {8..15,24..31}
    Frag16 a[2][8];
    #pragma unroll
    for (int m = 0; m < 2; ++m) {
        const unsigned short* rowp = xfB + (size_t)(rBase + m * 16 + l15) * EMB;
        #pragma unroll
        for (int cc = 0; cc < 8; ++cc) {
            const unsigned short* p = rowp + cc * 32 + hsel * 8;
            a[m][cc].q[0] = *(const v4u*)(p);
            a[m][cc].q[1] = *(const v4u*)(p + 16);
        }
    }

    float minv[2][8];
    int   mini[2][8];
    #pragma unroll
    for (int m = 0; m < 2; ++m)
        #pragma unroll
        for (int v = 0; v < 8; ++v) { minv[m][v] = 3.4e38f; mini[m][v] = 0; }

    // Cooperative B-tile staging: 16 rows x 512B = 512 v4u; 2 per thread.
    const int id0  = tid * 2;
    const int tRow = id0 >> 5;                  // 0..15
    const int tCol = id0 & 31;                  // even
    v4u st0, st1;

    auto loadTile = [&](int kt) {
        const unsigned short* p = cbB + (size_t)(kt * 16 + tRow) * EMB + tCol * 8;
        st0 = *(const v4u*)(p);
        st1 = *(const v4u*)(p + 8);
    };
    auto storeTile = [&](int buf) {
        unsigned short* q = &ldsB[buf][tRow * LDS_STRIDE_BF + tCol * 8];
        *(v4u*)(q)     = st0;
        *(v4u*)(q + 8) = st1;
    };

    loadTile(0);
    storeTile(0);
    float cbsCur = cbSq[l15];                   // cbSq for tile 0, in flight early
    __syncthreads();

    for (int kt = 0; kt < KTILES; ++kt) {
        const int  cur  = kt & 1;
        const bool more = (kt + 1) < KTILES;
        if (more) loadTile(kt + 1);
        // cbSq for the NEXT tile: issued after staging loads, consumed next iter,
        // so this iter's min-update only waits on last iter's (long-landed) value.
        const int ktn = more ? (kt + 1) : kt;
        const float cbsNext = cbSq[ktn * 16 + l15];
        if (kt + 2 < KTILES)  // -> global_prefetch_b8
            __builtin_prefetch(cbB + (size_t)((kt + 2) * 16 + tRow) * EMB + tCol * 8, 0, 1);

        v8f acc0 = {0.f,0.f,0.f,0.f,0.f,0.f,0.f,0.f};
        v8f acc1 = {0.f,0.f,0.f,0.f,0.f,0.f,0.f,0.f};

        // Register-pipelined B fragments: load frag cc+1 while WMMAs run on cc,
        // so the compiler can wait on dscnt<=2 instead of draining per step.
        const unsigned short* bbase = &ldsB[cur][l15 * LDS_STRIDE_BF + hsel * 8];
        Frag16 bc, bn;
        bc.q[0] = *(const v4u*)(bbase);
        bc.q[1] = *(const v4u*)(bbase + 16);
        #pragma unroll
        for (int cc = 0; cc < 8; ++cc) {
            if (cc < 7) {
                const unsigned short* bp = bbase + (cc + 1) * 32;
                bn.q[0] = *(const v4u*)(bp);
                bn.q[1] = *(const v4u*)(bp + 16);
            }
            acc0 = __builtin_amdgcn_wmma_f32_16x16x32_bf16(
                       false, a[0][cc].v, false, bc.v, (short)0, acc0, false, false);
            acc1 = __builtin_amdgcn_wmma_f32_16x16x32_bf16(
                       false, a[1][cc].v, false, bc.v, (short)0, acc1, false, false);
            if (cc < 7) bc = bn;
        }

        // C/D layout: this lane's column is k = kt*16 + (lane&15); 8 rows/VGPR.
        const int kcol = kt * 16 + l15;
        #pragma unroll
        for (int v = 0; v < 8; ++v) {
            const float s0 = fmaf(-2.f, acc0[v], cbsCur);
            if (s0 < minv[0][v]) { minv[0][v] = s0; mini[0][v] = kcol; }
            const float s1 = fmaf(-2.f, acc1[v], cbsCur);
            if (s1 < minv[1][v]) { minv[1][v] = s1; mini[1][v] = kcol; }
        }

        if (more) storeTile((kt + 1) & 1);
        cbsCur = cbsNext;
        __syncthreads();
    }

    // Cross-lane argmin within each 16-lane half; first-occurrence tie-break.
    #pragma unroll
    for (int m = 0; m < 2; ++m) {
        #pragma unroll
        for (int v = 0; v < 8; ++v) {
            float bv = minv[m][v];
            int   bi = mini[m][v];
            #pragma unroll
            for (int off = 8; off; off >>= 1) {
                const float ov = __shfl_xor(bv, off, 32);
                const int   oi = __shfl_xor(bi, off, 32);
                if (ov < bv || (ov == bv && oi < bi)) { bv = ov; bi = oi; }
            }
            if (l15 == 0) {
                const int row = v + hsel * 8;   // lanes 0-15: M=v; 16-31: M=v+8
                idxOut[rBase + m * 16 + row] = bi;
            }
        }
    }
}

// ---------------------------------------------------------------------------
// Kernel 3: gather fp32 codebook rows back to NCHW + indices (as float)
// grid = (32 h, 32 b), block = (32 w, 8)
// ---------------------------------------------------------------------------
__global__ __launch_bounds__(256)
void gather_kernel(const float* __restrict__ cb,
                   const int* __restrict__ idx,
                   float* __restrict__ outQ,
                   float* __restrict__ outIdxF) {
    const int tx = threadIdx.x;          // w
    const int ty = threadIdx.y;          // 0..7
    const int h  = blockIdx.x;
    const int b  = blockIdx.y;
    const int n  = b * 1024 + h * 32 + tx;
    const int k  = idx[n];
    if (ty == 0) outIdxF[n] = (float)k;
    const float* crow = cb + (size_t)k * EMB;
    #pragma unroll 4
    for (int c = ty; c < EMB; c += 8) {
        outQ[(((size_t)b * EMB + c) << 10) + h * 32 + tx] = crow[c]; // coalesced in w
    }
}

// ---------------------------------------------------------------------------
extern "C" void kernel_launch(void* const* d_in, const int* in_sizes, int n_in,
                              void* d_out, int out_size, void* d_ws, size_t ws_size,
                              hipStream_t stream) {
    const float* x  = (const float*)d_in[0];   // [32,256,32,32] f32
    const float* cb = (const float*)d_in[1];   // [8192,256] f32
    float* out = (float*)d_out;                // [8388608 quantized | 32768 index floats]

    // Workspace layout (~21 MB)
    char* ws = (char*)d_ws;
    unsigned short* xfB  = (unsigned short*)(ws);                       // 16 MB
    unsigned short* cbB  = (unsigned short*)(ws + (size_t)16777216);    //  4 MB
    float*          cbSq = (float*)         (ws + (size_t)20971520);    // 32 KB
    int*            idxW = (int*)           (ws + (size_t)21004288);    // 128 KB

    cb_prep_kernel    <<<NUM_EMBS, 256, 0, stream>>>(cb, cbB, cbSq);
    x_transpose_kernel<<<dim3(8, 32, 32), dim3(32, 32), 0, stream>>>(x, xfB);
    vq_argmin_kernel  <<<128, 256, 0, stream>>>(xfB, cbB, cbSq, idxW);
    gather_kernel     <<<dim3(32, 32), dim3(32, 8), 0, stream>>>(cb, idxW, out, out + QOUT);
}
